// Grid2Mesh_37288906064585
// MI455X (gfx1250) — matrix operand order repaired and checked by reading, not compiled, not run on hardware
//
#include <hip/hip_runtime.h>

// ---------------------------------------------------------------------------
// GraphCast Grid2Mesh block on CDNA5 (gfx1250).
// All GEMMs via v_wmma_f32_16x16x32_bf16 (wave32), bf16 operands / f32 acc.
// M=64 rows per block (4 row-tiles x 4 n-tiles per wave) -> 64 FLOP per L2
// weight byte; A staged in 64KB LDS K-chunks; 8 B-loads in flight per k-step.
// ---------------------------------------------------------------------------

typedef __attribute__((ext_vector_type(16))) __bf16 v16bf;
typedef __attribute__((ext_vector_type(8)))  __bf16 v8bf;
typedef __attribute__((ext_vector_type(8)))  float  v8f;
typedef __attribute__((ext_vector_type(4)))  unsigned short us4;

#define DIM 512
#define NGRID 131072
#define NMESH 40962
#define NEDGE 131072
#define BM 64            // rows per block

#if __has_builtin(__builtin_amdgcn_cvt_pk_bf16_f32)
#define HAVE_PK_BF16 1
typedef __attribute__((ext_vector_type(2))) __bf16 v2bf;
__device__ __forceinline__ unsigned int cvt2(float a, float b) {
  v2bf p = __builtin_amdgcn_cvt_pk_bf16_f32(a, b);
  union { v2bf v; unsigned int u; } c; c.v = p; return c.u;
}
#endif

__device__ __forceinline__ unsigned short f2bf(float f) {
#ifdef HAVE_PK_BF16
  return (unsigned short)(cvt2(f, f) & 0xffffu);
#else
  unsigned int u = __float_as_uint(f);
  u += 0x7fffu + ((u >> 16) & 1u);          // round-to-nearest-even
  return (unsigned short)(u >> 16);
#endif
}

__device__ __forceinline__ void store_pk4(unsigned short* dst, float4 f) {
#ifdef HAVE_PK_BF16
  union { us4 v; unsigned int u[2]; } c;
  c.u[0] = cvt2(f.x, f.y);
  c.u[1] = cvt2(f.z, f.w);
  *reinterpret_cast<us4*>(dst) = c.v;
#else
  us4 o; o.x = f2bf(f.x); o.y = f2bf(f.y); o.z = f2bf(f.z); o.w = f2bf(f.w);
  *reinterpret_cast<us4*>(dst) = o;
#endif
}

union AFrag { v16bf v; v8bf h[2]; };

// A operand (16x32 bf16, M x K): lane L<16 -> row L, K {0..15};
// lane L>=16 -> row L-16, K {16..31}.  Two 16B LDS loads.
__device__ __forceinline__ v16bf load_afrag(const unsigned short* base) {
  AFrag af;
  af.h[0] = *reinterpret_cast<const v8bf*>(base);
  af.h[1] = *reinterpret_cast<const v8bf*>(base + 8);
  return af.v;
}

// One K=512 chunk of GEMM accumulate.
// sA: [64 rows][512] bf16 in LDS.  Wt: [512 cols][KW] bf16 (pre-transposed).
// Each wave: 4 row-tiles x 4 n-tiles -> 16 WMMAs per k-step.  All 4 B
// fragments (8 global_load_b128) are issued before the WMMA burst for MLP.
template <int KW>
__device__ __forceinline__ void gemm_chunk(const unsigned short* sA,
                                           const unsigned short* __restrict__ Wt,
                                           int kBase, int wave, int hf, int ln,
                                           v8f acc[4][4]) {
  for (int kt = 0; kt < 16; ++kt) {
    const int kbA = kt * 32 + hf * 16;
    v16bf a[4];
#pragma unroll
    for (int rt = 0; rt < 4; ++rt)
      a[rt] = load_afrag(&sA[(rt * 16 + ln) * 512 + kbA]);
    const int kbB = kBase + kt * 32 + hf * 16;
    v16bf b[4];
#pragma unroll
    for (int j = 0; j < 4; ++j) {
      const int col = (wave * 4 + j) * 16 + ln;
      b[j] = *reinterpret_cast<const v16bf*>(&Wt[(long)col * KW + kbB]);
    }
#pragma unroll
    for (int j = 0; j < 4; ++j)
#pragma unroll
      for (int rt = 0; rt < 4; ++rt)
        acc[rt][j] = __builtin_amdgcn_wmma_f32_16x16x32_bf16(
            false, a[rt], false, b[j], (short)0, acc[rt][j], false, false);
  }
}

__device__ __forceinline__ void zero_acc(v8f acc[4][4]) {
  const v8f z = {0.f, 0.f, 0.f, 0.f, 0.f, 0.f, 0.f, 0.f};
#pragma unroll
  for (int rt = 0; rt < 4; ++rt)
#pragma unroll
    for (int j = 0; j < 4; ++j) acc[rt][j] = z;
}

// bias + ReLU the 64x512 hidden tile back into sA (as bf16), reset acc.
__device__ __forceinline__ void relu_stage(unsigned short* sA,
                                           const float* __restrict__ b1,
                                           int wave, int hf, int ln,
                                           v8f acc[4][4]) {
  const v8f z = {0.f, 0.f, 0.f, 0.f, 0.f, 0.f, 0.f, 0.f};
#pragma unroll
  for (int j = 0; j < 4; ++j) {
    const int col = (wave * 4 + j) * 16 + ln;
    const float bb = b1[col];
#pragma unroll
    for (int rt = 0; rt < 4; ++rt) {
#pragma unroll
      for (int v = 0; v < 8; ++v) {
        const int row = rt * 16 + v + hf * 8;
        float hv = acc[rt][j][v] + bb;
        sA[row * 512 + col] = f2bf(hv > 0.f ? hv : 0.f);
      }
      acc[rt][j] = z;
    }
  }
}

// ---------------------------------------------------------------------------
// Prep kernels
// ---------------------------------------------------------------------------
__global__ void zero_f32_kernel(float* __restrict__ p, long n) {
  long i = (long)blockIdx.x * 256 + threadIdx.x;
  if (i < n) p[i] = 0.0f;
}

// W: [K][N] fp32 row-major  ->  Wt: [N][K] bf16 row-major
__global__ void cvt_transpose_kernel(const float* __restrict__ W,
                                     unsigned short* __restrict__ Wt,
                                     int K, int N) {
  long i = (long)blockIdx.x * 256 + threadIdx.x;
  if (i < (long)K * N) {
    int k = (int)(i / N);
    int n = (int)(i % N);
    Wt[(long)n * K + k] = f2bf(W[i]);
  }
}

// ---------------------------------------------------------------------------
// Edge MLP:  h = ReLU([vg[src], vm[dst], e] @ W1 + b1);  upd = h @ W2 + b2
//            e_out = e + upd;  vm_agg[dst] += upd  (scatter-add)
// ---------------------------------------------------------------------------
__global__ __launch_bounds__(256)
void edge_kernel(const float* __restrict__ vg, const float* __restrict__ vm,
                 const float* __restrict__ eg,
                 const int* __restrict__ src_idx, const int* __restrict__ dst_idx,
                 const unsigned short* __restrict__ W1t, const float* __restrict__ b1,
                 const unsigned short* __restrict__ W2t, const float* __restrict__ b2,
                 float* __restrict__ e_out, float* __restrict__ vm_agg) {
  __shared__ __align__(32) unsigned short sA[BM * 512];   // 64 KB

  const int tid = threadIdx.x;
  const long e0 = (long)blockIdx.x * BM;
  const int wave = tid >> 5, lane = tid & 31;
  const int hf = lane >> 4, ln = lane & 15;

  v8f acc[4][4];
  zero_acc(acc);

  // Three K=512 segments of the concatenated feature: vg[src] | vm[dst] | e
  for (int chunk = 0; chunk < 3; ++chunk) {
    __syncthreads();
    const float* base = (chunk == 0) ? vg : (chunk == 1) ? vm : eg;
    for (int i = tid; i < BM * 128; i += 256) {
      const int r = i >> 7, c4 = (i & 127) << 2;
      long row;
      if (chunk == 0)      row = src_idx[e0 + r];
      else if (chunk == 1) row = dst_idx[e0 + r];
      else                 row = e0 + r;
      const float4 f = *reinterpret_cast<const float4*>(&base[row * DIM + c4]);
      store_pk4(&sA[r * 512 + c4], f);
    }
    __syncthreads();
    gemm_chunk<1536>(sA, W1t, chunk * 512, wave, hf, ln, acc);
  }

  __syncthreads();
  relu_stage(sA, b1, wave, hf, ln, acc);
  __syncthreads();

  gemm_chunk<512>(sA, W2t, 0, wave, hf, ln, acc);

  // epilogue: residual store + scatter-add into mesh aggregation buffer
#pragma unroll
  for (int rt = 0; rt < 4; ++rt) {
#pragma unroll
    for (int v = 0; v < 8; ++v) {
      const int rowl = rt * 16 + v + hf * 8;
      const long gr = e0 + rowl;
      const long dbase = (long)dst_idx[gr] * DIM;
#pragma unroll
      for (int j = 0; j < 4; ++j) {
        const int col = (wave * 4 + j) * 16 + ln;
        const float upd = acc[rt][j][v] + b2[col];
        e_out[gr * DIM + col] = eg[gr * DIM + col] + upd;
        atomicAdd(&vm_agg[dbase + col], upd);
      }
    }
  }
}

// ---------------------------------------------------------------------------
// Mesh node MLP:  upd = MLP([vm, vm_agg]);  vm_out = vm + upd
// ---------------------------------------------------------------------------
__global__ __launch_bounds__(256)
void mesh_kernel(const float* __restrict__ vm, const float* __restrict__ vm_agg,
                 const unsigned short* __restrict__ W1t, const float* __restrict__ b1,
                 const unsigned short* __restrict__ W2t, const float* __restrict__ b2,
                 float* __restrict__ vm_out) {
  __shared__ __align__(32) unsigned short sA[BM * 512];   // 64 KB

  const int tid = threadIdx.x;
  const long m0 = (long)blockIdx.x * BM;
  const int wave = tid >> 5, lane = tid & 31;
  const int hf = lane >> 4, ln = lane & 15;

  v8f acc[4][4];
  zero_acc(acc);

  for (int chunk = 0; chunk < 2; ++chunk) {
    __syncthreads();
    const float* base = (chunk == 0) ? vm : vm_agg;
    for (int i = tid; i < BM * 128; i += 256) {
      const int r = i >> 7, c4 = (i & 127) << 2;
      long m = m0 + r;
      if (m >= NMESH) m = NMESH - 1;              // tail clamp (loads only)
      const float4 f = *reinterpret_cast<const float4*>(&base[m * DIM + c4]);
      store_pk4(&sA[r * 512 + c4], f);
    }
    __syncthreads();
    gemm_chunk<1024>(sA, W1t, chunk * 512, wave, hf, ln, acc);
  }

  __syncthreads();
  relu_stage(sA, b1, wave, hf, ln, acc);
  __syncthreads();

  gemm_chunk<512>(sA, W2t, 0, wave, hf, ln, acc);

#pragma unroll
  for (int rt = 0; rt < 4; ++rt) {
#pragma unroll
    for (int v = 0; v < 8; ++v) {
      const int rowl = rt * 16 + v + hf * 8;
      const long m = m0 + rowl;
      if (m < NMESH) {
#pragma unroll
        for (int j = 0; j < 4; ++j) {
          const int col = (wave * 4 + j) * 16 + ln;
          vm_out[m * DIM + col] = vm[m * DIM + col] + acc[rt][j][v] + b2[col];
        }
      }
    }
  }
}

// ---------------------------------------------------------------------------
// Grid node MLP:  upd = MLP(vg);  vg_out = vg + upd
// ---------------------------------------------------------------------------
__global__ __launch_bounds__(256)
void grid_kernel(const float* __restrict__ vg,
                 const unsigned short* __restrict__ W1t, const float* __restrict__ b1,
                 const unsigned short* __restrict__ W2t, const float* __restrict__ b2,
                 float* __restrict__ vg_out) {
  __shared__ __align__(32) unsigned short sA[BM * 512];   // 64 KB

  const int tid = threadIdx.x;
  const long g0 = (long)blockIdx.x * BM;
  const int wave = tid >> 5, lane = tid & 31;
  const int hf = lane >> 4, ln = lane & 15;

  v8f acc[4][4];
  zero_acc(acc);

  for (int i = tid; i < BM * 128; i += 256) {
    const int r = i >> 7, c4 = (i & 127) << 2;
    const float4 f = *reinterpret_cast<const float4*>(&vg[(g0 + r) * DIM + c4]);
    store_pk4(&sA[r * 512 + c4], f);
  }
  __syncthreads();

  gemm_chunk<512>(sA, W1t, 0, wave, hf, ln, acc);

  __syncthreads();
  relu_stage(sA, b1, wave, hf, ln, acc);
  __syncthreads();

  gemm_chunk<512>(sA, W2t, 0, wave, hf, ln, acc);

#pragma unroll
  for (int rt = 0; rt < 4; ++rt) {
#pragma unroll
    for (int v = 0; v < 8; ++v) {
      const int rowl = rt * 16 + v + hf * 8;
      const long g = g0 + rowl;
#pragma unroll
      for (int j = 0; j < 4; ++j) {
        const int col = (wave * 4 + j) * 16 + ln;
        vg_out[g * DIM + col] = vg[g * DIM + col] + acc[rt][j][v] + b2[col];
      }
    }
  }
}

// ---------------------------------------------------------------------------
// Host launch
// ---------------------------------------------------------------------------
static inline size_t align256(size_t x) { return (x + 255) & ~(size_t)255; }

extern "C" void kernel_launch(void* const* d_in, const int* in_sizes, int n_in,
                              void* d_out, int out_size, void* d_ws, size_t ws_size,
                              hipStream_t stream) {
  const float* vg  = (const float*)d_in[0];
  const float* vm  = (const float*)d_in[1];
  const float* eg  = (const float*)d_in[2];
  const int*   src = (const int*)d_in[3];
  const int*   dst = (const int*)d_in[4];
  const float* We1 = (const float*)d_in[5];   const float* be1 = (const float*)d_in[6];
  const float* We2 = (const float*)d_in[7];   const float* be2 = (const float*)d_in[8];
  const float* Wm1 = (const float*)d_in[9];   const float* bm1 = (const float*)d_in[10];
  const float* Wm2 = (const float*)d_in[11];  const float* bm2 = (const float*)d_in[12];
  const float* Wg1 = (const float*)d_in[13];  const float* bg1 = (const float*)d_in[14];
  const float* Wg2 = (const float*)d_in[15];  const float* bg2 = (const float*)d_in[16];

  float* vg_out = (float*)d_out;
  float* vm_out = vg_out + (long)NGRID * DIM;
  float* e_out  = vm_out + (long)NMESH * DIM;

  // Workspace layout
  char* ws = (char*)d_ws;
  float* vm_agg = (float*)ws;                 ws += align256((size_t)NMESH * DIM * 4);
  unsigned short* We1t = (unsigned short*)ws; ws += align256((size_t)512 * 1536 * 2);
  unsigned short* We2t = (unsigned short*)ws; ws += align256((size_t)512 * 512 * 2);
  unsigned short* Wm1t = (unsigned short*)ws; ws += align256((size_t)512 * 1024 * 2);
  unsigned short* Wm2t = (unsigned short*)ws; ws += align256((size_t)512 * 512 * 2);
  unsigned short* Wg1t = (unsigned short*)ws; ws += align256((size_t)512 * 512 * 2);
  unsigned short* Wg2t = (unsigned short*)ws; ws += align256((size_t)512 * 512 * 2);

  // 1) zero the scatter-add accumulator
  const long nAgg = (long)NMESH * DIM;
  zero_f32_kernel<<<(int)((nAgg + 255) / 256), 256, 0, stream>>>(vm_agg, nAgg);

  // 2) weights -> bf16, transposed [N][K] for contiguous B-operand loads
  auto cvt = [&](const float* W, unsigned short* Wt, int K, int N) {
    long n = (long)K * N;
    cvt_transpose_kernel<<<(int)((n + 255) / 256), 256, 0, stream>>>(W, Wt, K, N);
  };
  cvt(We1, We1t, 1536, 512);
  cvt(We2, We2t, 512, 512);
  cvt(Wm1, Wm1t, 1024, 512);
  cvt(Wm2, Wm2t, 512, 512);
  cvt(Wg1, Wg1t, 512, 512);
  cvt(Wg2, Wg2t, 512, 512);

  // 3) edge MLP + residual + scatter-add  (must precede mesh kernel)
  edge_kernel<<<NEDGE / BM, 256, 0, stream>>>(vg, vm, eg, src, dst,
                                              We1t, be1, We2t, be2,
                                              e_out, vm_agg);

  // 4) mesh node MLP + residual
  mesh_kernel<<<(NMESH + BM - 1) / BM, 256, 0, stream>>>(vm, vm_agg,
                                                         Wm1t, bm1, Wm2t, bm2,
                                                         vm_out);

  // 5) grid node MLP + residual (independent of 3/4)
  grid_kernel<<<NGRID / BM, 256, 0, stream>>>(vg, Wg1t, bg1, Wg2t, bg2, vg_out);
}
